// MambaBlock_31980326486862
// MI455X (gfx1250) — matrix operand order repaired and checked by reading, not compiled, not run on hardware
//
#include <hip/hip_runtime.h>
#include <hip/hip_bf16.h>
#include <math.h>

typedef __attribute__((ext_vector_type(16))) __bf16 v16bf;
typedef __attribute__((ext_vector_type(8)))  float  v8f;
typedef __attribute__((ext_vector_type(4)))  unsigned u32x4;

namespace {
constexpr int kDModel = 768;
constexpr int kDState = 16;
constexpr int kDConv  = 4;
constexpr int kDIn    = 384;
constexpr int kDInner = 768;
constexpr int kDtRank = 24;
constexpr int kB      = 2;
constexpr int kL      = 2048;
constexpr int kTok    = kB * kL;   // 4096 tokens
constexpr float kEps  = 1e-5f;
// GEMM tiling
constexpr int BM = 128, BN = 64, BK = 32;
}

static __device__ __forceinline__ unsigned short f2bf(float f) {
  union { float f; unsigned u; } x; x.f = f;
  unsigned r = x.u + 0x7FFFu + ((x.u >> 16) & 1u);   // round-to-nearest-even
  return (unsigned short)(r >> 16);
}
static __device__ __forceinline__ float bf2f(unsigned short h) {
  union { unsigned u; float f; } x; x.u = ((unsigned)h) << 16;
  return x.f;
}
static __device__ __forceinline__ float sigmoidf_(float x) {
  return 1.0f / (1.0f + __expf(-x));
}

// ---------------------------------------------------------------------------
// CDNA5 async global->LDS copy (ASYNCcnt-tracked), with sync fallback.
// Probe result: param0 = 'int __vector(4) __device__ *' (addrspace(1), non-
// const), 4 args total.  Param1 is the LDS side (addrspace(3)).
// ---------------------------------------------------------------------------
#if __has_builtin(__builtin_amdgcn_global_load_async_to_lds_b128)
#define HAVE_ASYNC_LDS 1
#else
#define HAVE_ASYNC_LDS 0
#endif

typedef int v4i_ __attribute__((vector_size(16)));

static __device__ __forceinline__ void cp_g2l_16(void* lds_ptr, const void* gbl_ptr) {
#if HAVE_ASYNC_LDS
  __builtin_amdgcn_global_load_async_to_lds_b128(
      (__attribute__((address_space(1))) v4i_*)(unsigned long long)(uintptr_t)gbl_ptr,
      (__attribute__((address_space(3))) v4i_*)(unsigned)(uintptr_t)lds_ptr,
      0, 0);
#else
  *(u32x4*)lds_ptr = *(const u32x4*)gbl_ptr;
#endif
}
static __device__ __forceinline__ void wait_async_le6() {
#if HAVE_ASYNC_LDS
#if __has_builtin(__builtin_amdgcn_s_wait_asynccnt)
  __builtin_amdgcn_s_wait_asynccnt(6);
#else
  asm volatile("s_wait_asynccnt 0x6" ::: "memory");
#endif
#endif
}
static __device__ __forceinline__ void wait_async_0() {
#if HAVE_ASYNC_LDS
#if __has_builtin(__builtin_amdgcn_s_wait_asynccnt)
  __builtin_amdgcn_s_wait_asynccnt(0);
#else
  asm volatile("s_wait_asynccnt 0x0" ::: "memory");
#endif
#endif
}

// ---------------------------------------------------------------------------
// Pack an f32 [K x N] row-major weight into bf16 [Np x Kp] (transposed,
// zero-padded) so WMMA B-fragments load as contiguous 32-bit pairs.
// ---------------------------------------------------------------------------
__global__ void k_pack_weight(const float* __restrict__ src, unsigned short* __restrict__ dst,
                              int K, int N, int Kp, int Np) {
  int idx = blockIdx.x * blockDim.x + threadIdx.x;
  if (idx >= Kp * Np) return;
  int n = idx / Kp, k = idx % Kp;
  float v = (k < K && n < N) ? src[(size_t)k * N + n] : 0.0f;
  dst[idx] = f2bf(v);
}

// ---------------------------------------------------------------------------
// LayerNorm over D_MODEL=768, one block (256 thr) per token, bf16 output.
// ---------------------------------------------------------------------------
__global__ void k_layernorm_bf16(const float* __restrict__ x, const float* __restrict__ w,
                                 const float* __restrict__ b, unsigned short* __restrict__ out) {
  __shared__ float red[256];
  int tok = blockIdx.x, tid = threadIdx.x;
  const float* row = x + (size_t)tok * kDModel;
  float v0 = row[tid], v1 = row[tid + 256], v2 = row[tid + 512];
  red[tid] = v0 + v1 + v2;
  __syncthreads();
  for (int o = 128; o > 0; o >>= 1) { if (tid < o) red[tid] += red[tid + o]; __syncthreads(); }
  float mean = red[0] * (1.0f / kDModel);
  __syncthreads();
  float d0 = v0 - mean, d1 = v1 - mean, d2 = v2 - mean;
  red[tid] = d0 * d0 + d1 * d1 + d2 * d2;
  __syncthreads();
  for (int o = 128; o > 0; o >>= 1) { if (tid < o) red[tid] += red[tid + o]; __syncthreads(); }
  float rstd = rsqrtf(red[0] * (1.0f / kDModel) + kEps);
  unsigned short* orow = out + (size_t)tok * kDModel;
  orow[tid]       = f2bf(d0 * rstd * w[tid]       + b[tid]);
  orow[tid + 256] = f2bf(d1 * rstd * w[tid + 256] + b[tid + 256]);
  orow[tid + 512] = f2bf(d2 * rstd * w[tid + 512] + b[tid + 512]);
}

// ---------------------------------------------------------------------------
// Fused residual-add + LayerNorm, f32 output to d_out.
// ---------------------------------------------------------------------------
__global__ void k_add_layernorm(const float* __restrict__ resid, const float* __restrict__ o,
                                const float* __restrict__ w, const float* __restrict__ b,
                                float* __restrict__ out) {
  __shared__ float red[256];
  int tok = blockIdx.x, tid = threadIdx.x;
  size_t base = (size_t)tok * kDModel;
  float v0 = resid[base + tid]       + o[base + tid];
  float v1 = resid[base + tid + 256] + o[base + tid + 256];
  float v2 = resid[base + tid + 512] + o[base + tid + 512];
  red[tid] = v0 + v1 + v2;
  __syncthreads();
  for (int s = 128; s > 0; s >>= 1) { if (tid < s) red[tid] += red[tid + s]; __syncthreads(); }
  float mean = red[0] * (1.0f / kDModel);
  __syncthreads();
  float d0 = v0 - mean, d1 = v1 - mean, d2 = v2 - mean;
  red[tid] = d0 * d0 + d1 * d1 + d2 * d2;
  __syncthreads();
  for (int s = 128; s > 0; s >>= 1) { if (tid < s) red[tid] += red[tid + s]; __syncthreads(); }
  float rstd = rsqrtf(red[0] * (1.0f / kDModel) + kEps);
  out[base + tid]       = d0 * rstd * w[tid]       + b[tid];
  out[base + tid + 256] = d1 * rstd * w[tid + 256] + b[tid + 256];
  out[base + tid + 512] = d2 * rstd * w[tid + 512] + b[tid + 512];
}

// ---------------------------------------------------------------------------
// WMMA bf16 GEMM with double-buffered async global->LDS staging.
//   C[M x Np] = A[M x K] @ Wt^T (+bias, mode 1 = softplus epilogue)
//   A  : bf16 row-major, lda = K (K mult of 32); Wt : bf16 [Np x K]
// Block = 128 thr (4 waves), tile 128x64.  Wave: 32 rows x 64 cols
// (2 A-frags x 4 B-frags = 8 WMMA per K-step, B frags reused).
// LDS: double-buffered 128x32 A tile + 64x32 B tile = 24 KB.
// ---------------------------------------------------------------------------
__global__ __launch_bounds__(128)
void k_gemm_bf16(const unsigned short* __restrict__ A,
                 const unsigned short* __restrict__ Wt,
                 const float* __restrict__ bias,
                 float* __restrict__ C,
                 int K, int Np, int mode) {
  __shared__ unsigned short sA[2][BM * BK];   // [row][k]
  __shared__ unsigned short sB[2][BN * BK];   // [n][k]

  const int tid  = threadIdx.x;
  const int lane = tid & 31;
  const int wave = tid >> 5;
  const int gm0  = blockIdx.y * BM;
  const int n0   = blockIdx.x * BN;
  const int l16  = lane & 15;
  const int hi   = lane >> 4;

  const unsigned short* Ag = A  + (size_t)gm0 * K;
  const unsigned short* Wg = Wt + (size_t)n0 * K;

  const int rt = tid >> 2;          // 0..31
  const int pt = (tid & 3) * 8;     // element offset of 16B chunk within row

  union Frag { v16bf v; unsigned u[8]; };
  v8f acc[2][4];
#pragma unroll
  for (int a = 0; a < 2; ++a)
#pragma unroll
    for (int t = 0; t < 4; ++t) acc[a][t] = (v8f){0.f,0.f,0.f,0.f,0.f,0.f,0.f,0.f};

  auto issue = [&](int k0, int buf) {
#pragma unroll
    for (int i = 0; i < 4; ++i) {            // A tile: 128 rows
      int r = rt + 32 * i;
      cp_g2l_16(&sA[buf][r * BK + pt], Ag + (size_t)r * K + k0 + pt);
    }
#pragma unroll
    for (int i = 0; i < 2; ++i) {            // B tile: 64 rows (cols of C)
      int r = rt + 32 * i;
      cp_g2l_16(&sB[buf][r * BK + pt], Wg + (size_t)r * K + k0 + pt);
    }
  };

  const int nsteps = K / BK;
  issue(0, 0);
  for (int s = 0; s < nsteps; ++s) {
    const int buf = s & 1;
    if (s + 1 < nsteps) { issue((s + 1) * BK, buf ^ 1); wait_async_le6(); }
    else                { wait_async_0(); }
    __syncthreads();

    // A fragments from LDS (16x32 bf16): VGPR j pair at k=(j>>2)*16+hi*8+(j&3)*2
    const unsigned short* sa = &sA[buf][(wave * 32 + l16) * BK];
    Frag a0, a1;
#pragma unroll
    for (int j = 0; j < 8; ++j) {
      int k = ((j >> 2) << 4) + (hi << 3) + ((j & 3) << 1);
      a0.u[j] = *(const unsigned*)(sa + k);
      a1.u[j] = *(const unsigned*)(sa + 16 * BK + k);
    }
#pragma unroll
    for (int t = 0; t < 4; ++t) {
      // B fragment (32x16 bf16): lane col n, VGPR j pair at k=(j>>2)*16+hi*8+(j&3)*2
      const unsigned short* bc = &sB[buf][(t * 16 + l16) * BK + (hi << 3)];
      Frag bf;
#pragma unroll
      for (int j = 0; j < 8; ++j) bf.u[j] = *(const unsigned*)(bc + ((j >> 2) << 4) + ((j & 3) << 1));
      acc[0][t] = __builtin_amdgcn_wmma_f32_16x16x32_bf16(
          false, a0.v, false, bf.v, (short)0, acc[0][t], false, false);
      acc[1][t] = __builtin_amdgcn_wmma_f32_16x16x32_bf16(
          false, a1.v, false, bf.v, (short)0, acc[1][t], false, false);
    }
    __syncthreads();
  }

  // Epilogue: C/D layout — VGPR r: m = base + hi*8 + r, n = n0 + t*16 + l16
#pragma unroll
  for (int a = 0; a < 2; ++a) {
#pragma unroll
    for (int t = 0; t < 4; ++t) {
      int n = n0 + t * 16 + l16;
      float bv = bias ? bias[n] : 0.0f;
#pragma unroll
      for (int r = 0; r < 8; ++r) {
        int m = gm0 + wave * 32 + a * 16 + (hi << 3) + r;
        float v = acc[a][t][r] + bv;
        if (mode == 1) v = (v > 20.0f) ? v : log1pf(__expf(v));
        C[(size_t)m * Np + n] = v;
      }
    }
  }
}

// ---------------------------------------------------------------------------
// Split xp -> per-direction bf16 input, flipping time for the backward dir.
// ---------------------------------------------------------------------------
__global__ void k_split_u(const float* __restrict__ xp, unsigned short* __restrict__ u, int dir) {
  int idx = blockIdx.x * blockDim.x + threadIdx.x;
  if (idx >= kTok * kDIn) return;
  int c = idx % kDIn, tok = idx / kDIn;
  int b = tok / kL, t = tok % kL;
  int tsrc = dir ? (kL - 1 - t) : t;
  u[idx] = f2bf(xp[(size_t)(b * kL + tsrc) * (2 * kDIn) + dir * kDIn + c]);
}

// ---------------------------------------------------------------------------
// Causal depthwise conv (K=4) over the x-half of xz, + SiLU, bf16 out.
// ---------------------------------------------------------------------------
__global__ void k_conv_silu(const float* __restrict__ xz, const float* __restrict__ cw,
                            const float* __restrict__ cb, unsigned short* __restrict__ xc) {
  int idx = blockIdx.x * blockDim.x + threadIdx.x;
  if (idx >= kTok * kDInner) return;
  int d = idx % kDInner, tok = idx / kDInner;
  int b = tok / kL, t = tok % kL;
  float acc = cb[d];
#pragma unroll
  for (int k = 0; k < kDConv; ++k) {
    int ts = t - (kDConv - 1) + k;
    if (ts >= 0) acc += xz[(size_t)(b * kL + ts) * (2 * kDInner) + d] * cw[d * kDConv + k];
  }
  xc[idx] = f2bf(acc * sigmoidf_(acc));
}

// ---------------------------------------------------------------------------
// Pack dt-rank columns of xd (f32, ld=64) into bf16 [tok x 32] (K padded).
// ---------------------------------------------------------------------------
__global__ void k_pack_dt_in(const float* __restrict__ xd, unsigned short* __restrict__ adt) {
  int idx = blockIdx.x * blockDim.x + threadIdx.x;
  if (idx >= kTok * 32) return;
  int c = idx % 32, tok = idx / 32;
  adt[idx] = f2bf(c < kDtRank ? xd[(size_t)tok * 64 + c] : 0.0f);
}

// ---------------------------------------------------------------------------
// Selective scan, (channel x state)-parallel: 16 lanes per channel, one
// state element h per lane -> 1 exp per step.  y = sum_n(h*C) via 4
// __shfl_xor steps within the 16-lane group (wave32, no LDS, no barriers).
// grid = (B, D_INNER/16), block = 256 (16 channels / 8 waves per block).
// ---------------------------------------------------------------------------
__global__ __launch_bounds__(256)
void k_scan(const float* __restrict__ dt, const unsigned short* __restrict__ xc,
            const float* __restrict__ xd, const float* __restrict__ xz,
            const float* __restrict__ A_log, const float* __restrict__ Dp,
            unsigned short* __restrict__ ybf) {
  const int b = blockIdx.x;
  const int n = threadIdx.x & 15;                       // state index
  const int d = blockIdx.y * 16 + (threadIdx.x >> 4);   // channel
  float A  = -__expf(A_log[(size_t)d * kDState + n]);
  float h  = 0.0f;
  float Dd = Dp[d];
  for (int t = 0; t < kL; ++t) {
    size_t tok = (size_t)b * kL + t;
    float dtv = dt[tok * kDInner + d];
    float xv  = bf2f(xc[tok * kDInner + d]);
    float Bv  = xd[tok * 64 + kDtRank + n];
    float Cv  = xd[tok * 64 + kDtRank + kDState + n];
    h = __expf(dtv * A) * h + dtv * xv * Bv;
    float p = h * Cv;
    p += __shfl_xor(p, 1, 16);
    p += __shfl_xor(p, 2, 16);
    p += __shfl_xor(p, 4, 16);
    p += __shfl_xor(p, 8, 16);
    if (n == 0) {
      float z = xz[tok * (2 * kDInner) + kDInner + d];
      float y = (p + xv * Dd) * (z * sigmoidf_(z));
      ybf[tok * kDInner + d] = f2bf(y);
    }
  }
}

// ---------------------------------------------------------------------------
// Place per-direction out-proj result into the concat buffer (bf16), flipping
// time back for the backward direction.
// ---------------------------------------------------------------------------
__global__ void k_place(const float* __restrict__ src, unsigned short* __restrict__ dst, int dir) {
  int idx = blockIdx.x * blockDim.x + threadIdx.x;
  if (idx >= kTok * kDIn) return;
  int c = idx % kDIn, tok = idx / kDIn;
  int b = tok / kL, t = tok % kL;
  int tdst = dir ? (kL - 1 - t) : t;
  dst[(size_t)(b * kL + tdst) * (2 * kDIn) + dir * kDIn + c] = f2bf(src[idx]);
}

// ===========================================================================
extern "C" void kernel_launch(void* const* d_in, const int* in_sizes, int n_in,
                              void* d_out, int out_size, void* d_ws, size_t ws_size,
                              hipStream_t stream) {
  (void)in_sizes; (void)n_in; (void)out_size; (void)ws_size;
  const float* F[31];
  for (int i = 0; i < 31; ++i) F[i] = (const float*)d_in[i];
  const float* x      = F[0];
  const float* innw   = F[1];
  const float* innb   = F[2];
  const float* ipw    = F[3];
  const float* ipb    = F[4];
  const float* opw    = F[27];
  const float* opb    = F[28];
  const float* normw  = F[29];
  const float* normb  = F[30];
  // per-direction params: in_w,in_b,conv_w,conv_b,xp_w,dt_w,dt_b,A_log,D,out_w,out_b
  const float* P[2][11];
  for (int dir = 0; dir < 2; ++dir)
    for (int j = 0; j < 11; ++j) P[dir][j] = F[5 + dir * 11 + j];

  // ---- workspace bump allocator -------------------------------------------
  char* ws = (char*)d_ws;
  size_t off = 0;
  auto alloc = [&](size_t bytes) -> void* {
    void* p = ws + off;
    off += (bytes + 255) & ~(size_t)255;
    return p;
  };
  unsigned short* wt_ip     = (unsigned short*)alloc((size_t)768 * 768 * 2);
  unsigned short* wt_in[2]  = { (unsigned short*)alloc((size_t)1536 * 384 * 2),
                                (unsigned short*)alloc((size_t)1536 * 384 * 2) };
  unsigned short* wt_xp[2]  = { (unsigned short*)alloc((size_t)64 * 768 * 2),
                                (unsigned short*)alloc((size_t)64 * 768 * 2) };
  unsigned short* wt_dt[2]  = { (unsigned short*)alloc((size_t)768 * 32 * 2),
                                (unsigned short*)alloc((size_t)768 * 32 * 2) };
  unsigned short* wt_out[2] = { (unsigned short*)alloc((size_t)384 * 768 * 2),
                                (unsigned short*)alloc((size_t)384 * 768 * 2) };
  unsigned short* wt_op     = (unsigned short*)alloc((size_t)768 * 768 * 2);
  unsigned short* xn_bf     = (unsigned short*)alloc((size_t)kTok * 768 * 2);
  float*          xp_f      = (float*)alloc((size_t)kTok * 768 * 4);
  unsigned short* u_bf[2]   = { (unsigned short*)alloc((size_t)kTok * 384 * 2),
                                (unsigned short*)alloc((size_t)kTok * 384 * 2) };
  float*          xz_f      = (float*)alloc((size_t)kTok * 1536 * 4);
  unsigned short* xc_bf     = (unsigned short*)alloc((size_t)kTok * 768 * 2);
  float*          xd_f      = (float*)alloc((size_t)kTok * 64 * 4);
  unsigned short* adt_bf    = (unsigned short*)alloc((size_t)kTok * 32 * 2);
  float*          dt_f      = (float*)alloc((size_t)kTok * 768 * 4);
  unsigned short* y_bf      = (unsigned short*)alloc((size_t)kTok * 768 * 2);
  float*          od_f      = (float*)alloc((size_t)kTok * 384 * 4);
  unsigned short* cat_bf    = (unsigned short*)alloc((size_t)kTok * 768 * 2);
  float*          o_f       = (float*)alloc((size_t)kTok * 768 * 4);

  auto blocks = [](long n) { return dim3((unsigned)((n + 255) / 256)); };

  // ---- weight packing (f32 -> bf16, transposed + padded) ------------------
  k_pack_weight<<<blocks(768 * 768), 256, 0, stream>>>(ipw, wt_ip, 768, 768, 768, 768);
  for (int dir = 0; dir < 2; ++dir) {
    k_pack_weight<<<blocks(1536 * 384), 256, 0, stream>>>(P[dir][0], wt_in[dir], 384, 1536, 384, 1536);
    k_pack_weight<<<blocks(64 * 768),   256, 0, stream>>>(P[dir][4], wt_xp[dir], 768, 56, 768, 64);
    k_pack_weight<<<blocks(768 * 32),   256, 0, stream>>>(P[dir][5], wt_dt[dir], 24, 768, 32, 768);
    k_pack_weight<<<blocks(384 * 768),  256, 0, stream>>>(P[dir][9], wt_out[dir], 768, 384, 768, 384);
  }
  k_pack_weight<<<blocks(768 * 768), 256, 0, stream>>>(opw, wt_op, 768, 768, 768, 768);

  // ---- input layernorm + in-proj GEMM -------------------------------------
  k_layernorm_bf16<<<kTok, 256, 0, stream>>>(x, innw, innb, xn_bf);
  k_gemm_bf16<<<dim3(768 / BN, kTok / BM), 128, 0, stream>>>(xn_bf, wt_ip, ipb, xp_f, 768, 768, 0);
  k_split_u<<<blocks((long)kTok * kDIn), 256, 0, stream>>>(xp_f, u_bf[0], 0);
  k_split_u<<<blocks((long)kTok * kDIn), 256, 0, stream>>>(xp_f, u_bf[1], 1);

  // ---- per-direction Mamba SSM --------------------------------------------
  for (int dir = 0; dir < 2; ++dir) {
    k_gemm_bf16<<<dim3(1536 / BN, kTok / BM), 128, 0, stream>>>(
        u_bf[dir], wt_in[dir], P[dir][1], xz_f, 384, 1536, 0);
    k_conv_silu<<<blocks((long)kTok * kDInner), 256, 0, stream>>>(
        xz_f, P[dir][2], P[dir][3], xc_bf);
    k_gemm_bf16<<<dim3(64 / BN, kTok / BM), 128, 0, stream>>>(
        xc_bf, wt_xp[dir], nullptr, xd_f, 768, 64, 0);
    k_pack_dt_in<<<blocks((long)kTok * 32), 256, 0, stream>>>(xd_f, adt_bf);
    k_gemm_bf16<<<dim3(768 / BN, kTok / BM), 128, 0, stream>>>(
        adt_bf, wt_dt[dir], P[dir][6], dt_f, 32, 768, 1);   // softplus epilogue
    k_scan<<<dim3(kB, kDInner / 16), 256, 0, stream>>>(
        dt_f, xc_bf, xd_f, xz_f, P[dir][7], P[dir][8], y_bf);
    k_gemm_bf16<<<dim3(384 / BN, kTok / BM), 128, 0, stream>>>(
        y_bf, wt_out[dir], P[dir][10], od_f, 768, 384, 0);
    k_place<<<blocks((long)kTok * kDIn), 256, 0, stream>>>(od_f, cat_bf, dir);
  }

  // ---- final projection + residual layernorm ------------------------------
  k_gemm_bf16<<<dim3(768 / BN, kTok / BM), 128, 0, stream>>>(cat_bf, wt_op, opb, o_f, 768, 768, 0);
  k_add_layernorm<<<kTok, 256, 0, stream>>>(x, o_f, normw, normb, (float*)d_out);
}
